// DecoderBasicBlock_19301583028281
// MI455X (gfx1250) — matrix operand order repaired and checked by reading, not compiled, not run on hardware
//
#include <hip/hip_runtime.h>
#include <hip/hip_bf16.h>
#include <math.h>

typedef __attribute__((ext_vector_type(16))) _Float16 v16h;
typedef __attribute__((ext_vector_type(8)))  float    v8f;

#define DIMD   512
#define HEADS  8
#define HD     64
#define PTOK   10
#define BATCH  8
#define NXTOK  2048
#define NXROWS (BATCH*NXTOK)   // 16384
#define NMROWS (BATCH*PTOK)    // 80
#define LSEQ   522
#define NSEQ   64              // B * V
#define LPSEQ  512

__device__ __forceinline__ float gelu_f(float x){
  return 0.5f * x * (1.0f + erff(x * 0.70710678118654752f));
}
__device__ __forceinline__ float waveMax(float v){
  #pragma unroll
  for (int m = 16; m > 0; m >>= 1) v = fmaxf(v, __shfl_xor(v, m, 32));
  return v;
}
__device__ __forceinline__ float waveSum(float v){
  #pragma unroll
  for (int m = 16; m > 0; m >>= 1) v += __shfl_xor(v, m, 32);
  return v;
}

// ---------------------------------------------------------------------------
// WMMA GEMM: C[M,N] = act(A[M,K] @ W[K,N] + bias) + resid
//
// Block = 256 threads (8 waves) computing a 128(M) x 64(N) tile.
// All 8 waves share one N-strip: the 32(K) x 64(N) B block is staged through
// LDS once per block (coalesced global_load_b128, f32->f16 convert, scattered
// into fragment order), double-buffered, so each wave's B fragment read is a
// contiguous 32-byte ds_load. A fragments are per-wave float4 loads.
// Requirements: M % 16 == 0, N % 64 == 0, K % 32 == 0 (true at all call sites).
// ---------------------------------------------------------------------------
template<int BIAS, int ACT, int RES>
__global__ __launch_bounds__(256) void gemm_f16wmma(
    const float* __restrict__ A, const float* __restrict__ W,
    const float* __restrict__ bias, const float* __restrict__ resid,
    float* __restrict__ C, int M, int N, int K)
{
  const int mTiles  = M >> 4;
  const int mBlocks = (mTiles + 7) >> 3;
  const int mb  = blockIdx.x % mBlocks;
  const int tn  = blockIdx.x / mBlocks;
  const int wid  = threadIdx.x >> 5;
  const int lane = threadIdx.x & 31;
  const int half = lane >> 4;      // 0: lanes 0-15, 1: lanes 16-31
  const int l16  = lane & 15;
  const int tm   = mb * 8 + wid;
  const bool active = (tm < mTiles);   // wave-uniform; EXEC stays all-ones
  const int colBase = tn * 64;

  // Fragment-ordered B staging: sB[buf][j*512 + lane*16 + i]
  // (j = 16-col subtile 0..3, lane = half*16 + n%16, i -> K = kk + half*16 + i)
  __shared__ _Float16 sB[2][2048];

  const float* arow = A + (size_t)(tm * 16 + l16) * K;   // deref only if active

  v8f acc0 = {}, acc1 = {}, acc2 = {}, acc3 = {};

  auto loadB = [&](int kk, int buf){
    #pragma unroll
    for (int e = 0; e < 2; e++){
      int idx = threadIdx.x + e * 256;   // 512 float4 = 32 rows x 64 cols
      int row = idx >> 4;                // K offset within block: 0..31
      int cg  = idx & 15;                // float4 group within the 64 cols
      const float4 src = *(const float4*)(W + (size_t)(kk + row) * N + colBase + cg * 4);
      float vals[4] = {src.x, src.y, src.z, src.w};
      int hi = row >> 4, ii = row & 15;
      #pragma unroll
      for (int e2 = 0; e2 < 4; e2++){
        int n  = cg * 4 + e2;
        int j  = n >> 4, nl = n & 15;
        sB[buf][j * 512 + (hi * 16 + nl) * 16 + ii] = (_Float16)vals[e2];
      }
    }
  };

  loadB(0, 0);
  __syncthreads();

  const int steps = K >> 5;
  for (int s = 0; s < steps; s++){
    int kk = s << 5;
    if (s + 1 < steps) loadB(kk + 32, (s + 1) & 1);

    if (active){
      // A fragment: lanes 0-15 K=kk+0..15 ; lanes 16-31 K=kk+16..31
      const float4* ap  = (const float4*)(arow + kk + half * 16);
      float4 a0 = ap[0], a1 = ap[1];
      const float4* ap2 = (const float4*)(arow + kk + half * 16 + 8);
      float4 a2 = ap2[0], a3 = ap2[1];
      v16h af;
      af[0]=(_Float16)a0.x;  af[1]=(_Float16)a0.y;  af[2]=(_Float16)a0.z;  af[3]=(_Float16)a0.w;
      af[4]=(_Float16)a1.x;  af[5]=(_Float16)a1.y;  af[6]=(_Float16)a1.z;  af[7]=(_Float16)a1.w;
      af[8]=(_Float16)a2.x;  af[9]=(_Float16)a2.y;  af[10]=(_Float16)a2.z; af[11]=(_Float16)a2.w;
      af[12]=(_Float16)a3.x; af[13]=(_Float16)a3.y; af[14]=(_Float16)a3.z; af[15]=(_Float16)a3.w;

      const v16h* bp = (const v16h*)&sB[s & 1][0];
      v16h bf0 = bp[0 * 32 + lane];
      v16h bf1 = bp[1 * 32 + lane];
      v16h bf2 = bp[2 * 32 + lane];
      v16h bf3 = bp[3 * 32 + lane];

      acc0 = __builtin_amdgcn_wmma_f32_16x16x32_f16(false, af, false, bf0, (short)0, acc0, false, false);
      acc1 = __builtin_amdgcn_wmma_f32_16x16x32_f16(false, af, false, bf1, (short)0, acc1, false, false);
      acc2 = __builtin_amdgcn_wmma_f32_16x16x32_f16(false, af, false, bf2, (short)0, acc2, false, false);
      acc3 = __builtin_amdgcn_wmma_f32_16x16x32_f16(false, af, false, bf3, (short)0, acc3, false, false);
    }
    __syncthreads();
  }

  if (active){
    int rowBase = tm * 16 + half * 8;  // C/D layout: VGPR r -> M = r + 8*half, N = lane%16
    int colB    = colBase + l16;
    #define EPILOG(ACC, J) { \
      int col = colB + (J) * 16; \
      float bv = BIAS ? bias[col] : 0.0f; \
      for (int r = 0; r < 8; r++){ \
        size_t idx = (size_t)(rowBase + r) * N + col; \
        float v = ACC[r] + bv; \
        if (ACT) v = gelu_f(v); \
        if (RES) v += resid[idx]; \
        C[idx] = v; \
      } }
    EPILOG(acc0, 0) EPILOG(acc1, 1) EPILOG(acc2, 2) EPILOG(acc3, 3)
    #undef EPILOG
  }
}

// ---------------------------------------------------------------------------
// LayerNorm over rows of width 512; one wave per row.
// ---------------------------------------------------------------------------
__global__ __launch_bounds__(256) void layernorm512(
    const float* __restrict__ X, const float* __restrict__ g, const float* __restrict__ b,
    float* __restrict__ Y, int R)
{
  int wid = blockIdx.x * 8 + (threadIdx.x >> 5);
  if (wid >= R) return;
  int lane = threadIdx.x & 31;
  const float* x = X + (size_t)wid * DIMD;
  float v[16]; float s = 0.f;
  #pragma unroll
  for (int i = 0; i < 16; i++){ v[i] = x[lane + i * 32]; s += v[i]; }
  s = waveSum(s);
  float mean = s * (1.0f / DIMD);
  float var = 0.f;
  #pragma unroll
  for (int i = 0; i < 16; i++){ float d = v[i] - mean; var += d * d; }
  var = waveSum(var);
  float rstd = rsqrtf(var * (1.0f / DIMD) + 1e-5f);
  float* y = Y + (size_t)wid * DIMD;
  #pragma unroll
  for (int i = 0; i < 16; i++){
    int c = lane + i * 32;
    y[c] = (v[i] - mean) * rstd * g[c] + b[c];
  }
}

// ---------------------------------------------------------------------------
// Attention, few queries (mem attends over x): one block per (b,h,q),
// softmax over 2048 keys staged in LDS.
// qh: (B, PTOK, 512) ; kvh: (B, NXTOK, 1024) [k | v] ; o: (B, PTOK, 512)
// ---------------------------------------------------------------------------
__global__ __launch_bounds__(256) void attn_qsmall(
    const float* __restrict__ qh, const float* __restrict__ kvh, float* __restrict__ o)
{
  int bid = blockIdx.x;
  int q = bid % PTOK;
  int h = (bid / PTOK) % HEADS;
  int b = bid / (PTOK * HEADS);
  __shared__ float sQ[HD];
  __shared__ float sAtt[NXTOK];
  __shared__ float sRed[8];
  __shared__ float sO[4][HD];
  int t = threadIdx.x;
  if (t < HD) sQ[t] = qh[((size_t)(b * PTOK + q)) * DIMD + h * HD + t];
  __syncthreads();

  float lmax = -3.0e38f;
  for (int k = t; k < NXTOK; k += 256){
    const float* kp = kvh + ((size_t)(b * NXTOK + k)) * 1024 + h * HD;
    float d = 0.f;
    #pragma unroll
    for (int i = 0; i < HD; i++) d += sQ[i] * kp[i];
    d *= 0.125f;                    // 1/sqrt(64)
    sAtt[k] = d;
    lmax = fmaxf(lmax, d);
  }
  lmax = waveMax(lmax);
  if ((t & 31) == 0) sRed[t >> 5] = lmax;
  __syncthreads();
  float bmax = sRed[0];
  #pragma unroll
  for (int i = 1; i < 8; i++) bmax = fmaxf(bmax, sRed[i]);
  __syncthreads();

  float lsum = 0.f;
  for (int k = t; k < NXTOK; k += 256){
    float e = expf(sAtt[k] - bmax);
    sAtt[k] = e;
    lsum += e;
  }
  lsum = waveSum(lsum);
  if ((t & 31) == 0) sRed[t >> 5] = lsum;
  __syncthreads();
  float bsum = 0.f;
  #pragma unroll
  for (int i = 0; i < 8; i++) bsum += sRed[i];
  float inv = 1.0f / bsum;

  int d = t & 63, part = t >> 6;    // 4 key-partitions x 64 dims
  float accu = 0.f;
  for (int k = part; k < NXTOK; k += 4)
    accu += sAtt[k] * kvh[((size_t)(b * NXTOK + k)) * 1024 + DIMD + h * HD + d];
  sO[part][d] = accu;
  __syncthreads();
  if (t < HD){
    float r = (sO[0][t] + sO[1][t] + sO[2][t] + sO[3][t]) * inv;
    o[((size_t)(b * PTOK + q)) * DIMD + h * HD + t] = r;
  }
}

// ---------------------------------------------------------------------------
// Attention, few keys (x attends over mem): thread per query row, 10 keys in LDS.
// qh: (B, NXTOK, 512) ; kvh: (B, PTOK, 1024) ; o: (B, NXTOK, 512)
// grid: (NXTOK/256, HEADS, BATCH)
// ---------------------------------------------------------------------------
__global__ __launch_bounds__(256) void attn_ksmall(
    const float* __restrict__ qh, const float* __restrict__ kvh, float* __restrict__ o)
{
  int b = blockIdx.z, h = blockIdx.y;
  int qi = blockIdx.x * 256 + threadIdx.x;
  __shared__ float sK[PTOK * HD], sV[PTOK * HD];
  for (int i = threadIdx.x; i < PTOK * HD; i += 256){
    int k = i >> 6, d = i & 63;
    sK[i] = kvh[((size_t)(b * PTOK + k)) * 1024 + h * HD + d];
    sV[i] = kvh[((size_t)(b * PTOK + k)) * 1024 + DIMD + h * HD + d];
  }
  __syncthreads();
  const float* qp = qh + ((size_t)(b * NXTOK + qi)) * DIMD + h * HD;
  float qv[HD];
  #pragma unroll
  for (int i = 0; i < HD; i++) qv[i] = qp[i];
  float att[PTOK]; float mx = -3.0e38f;
  #pragma unroll
  for (int k = 0; k < PTOK; k++){
    float d = 0.f;
    #pragma unroll
    for (int i = 0; i < HD; i++) d += qv[i] * sK[k * HD + i];
    d *= 0.125f; att[k] = d; mx = fmaxf(mx, d);
  }
  float s = 0.f;
  #pragma unroll
  for (int k = 0; k < PTOK; k++){ att[k] = expf(att[k] - mx); s += att[k]; }
  float inv = 1.0f / s;
  float ov[HD];
  #pragma unroll
  for (int i = 0; i < HD; i++) ov[i] = 0.f;
  #pragma unroll
  for (int k = 0; k < PTOK; k++){
    float w = att[k] * inv;
    #pragma unroll
    for (int i = 0; i < HD; i++) ov[i] += w * sV[k * HD + i];
  }
  float* op = o + ((size_t)(b * NXTOK + qi)) * DIMD + h * HD;
  #pragma unroll
  for (int i = 0; i < HD; i++) op[i] = ov[i];
}

// ---------------------------------------------------------------------------
// LRU scans. The FFT conv in the reference is y[t] = Re(g * u[t]),
// u[t] = x[t] + lam*u[t-1], lam = exp(-exp(nu))*(cos th + i sin th).
// Only the first PTOK outputs are consumed downstream, so the forward scan
// runs 10 steps; the backward scan runs the full 522 steps but stores 10.
// Z is never materialized: x[t] is read from mem/ts on the fly.
// Y (fwd -> cols 0..511, bwd -> cols 512..1023) is (NSEQ*PTOK, 1024).
// ---------------------------------------------------------------------------
__global__ __launch_bounds__(256) void lru_fwd_kernel(
    const float* __restrict__ mem, const float* __restrict__ prefix,
    const float* __restrict__ nu, const float* __restrict__ th,
    const float* __restrict__ gr, const float* __restrict__ gi,
    float* __restrict__ Y)
{
  int tid = blockIdx.x * 256 + threadIdx.x;
  if (tid >= NSEQ * DIMD) return;
  int d = tid & (DIMD - 1);
  int s = tid >> 9;
  int b = s >> 3;
  float mag = expf(-expf(nu[d]));
  float lr = mag * cosf(th[d]), li = mag * sinf(th[d]);
  float grv = gr[d], giv = gi[d], pe = prefix[d];
  float ur = 0.f, ui = 0.f;
  for (int t2 = 0; t2 < PTOK; t2++){
    float x = mem[((size_t)(b * PTOK + t2)) * DIMD + d] + pe;
    float nr = x + lr * ur - li * ui;
    float ni = li * ur + lr * ui;
    ur = nr; ui = ni;
    Y[((size_t)(s * PTOK + t2)) * 1024 + d] = grv * ur - giv * ui;
  }
}

__global__ __launch_bounds__(256) void lru_bwd_kernel(
    const float* __restrict__ mem, const float* __restrict__ ts,
    const float* __restrict__ prefix, const float* __restrict__ signal,
    const float* __restrict__ nu, const float* __restrict__ th,
    const float* __restrict__ gr, const float* __restrict__ gi,
    float* __restrict__ Y)
{
  int tid = blockIdx.x * 256 + threadIdx.x;
  if (tid >= NSEQ * DIMD) return;
  int d = tid & (DIMD - 1);
  int s = tid >> 9;
  int b = s >> 3;
  int v = s & 7;
  float mag = expf(-expf(nu[d]));
  float lr = mag * cosf(th[d]), li = mag * sinf(th[d]);
  float grv = gr[d], giv = gi[d], pe = prefix[d], se = signal[d];
  float ur = 0.f, ui = 0.f;
  for (int t2 = LSEQ - 1; t2 >= 0; t2--){
    float x;
    if (t2 >= PTOK)
      x = ts[(((size_t)b * LPSEQ + (t2 - PTOK)) * 8 + v) * DIMD + d] + se;
    else
      x = mem[((size_t)(b * PTOK + t2)) * DIMD + d] + pe;
    float nr = x + lr * ur - li * ui;
    float ni = li * ur + lr * ui;
    ur = nr; ui = ni;
    if (t2 < PTOK)
      Y[((size_t)(s * PTOK + t2)) * 1024 + DIMD + d] = grv * ur - giv * ui;
  }
}

// ---------------------------------------------------------------------------
// Glue kernels
// ---------------------------------------------------------------------------
__global__ __launch_bounds__(256) void split_x_kernel(
    const float* __restrict__ x, float* __restrict__ memin, float* __restrict__ xin)
{
  size_t tid = (size_t)blockIdx.x * 256 + threadIdx.x;
  const size_t total = (size_t)BATCH * (PTOK + NXTOK) * DIMD;
  if (tid >= total) return;
  int d = (int)(tid & (DIMD - 1));
  size_t r = tid >> 9;
  int row = (int)(r % (PTOK + NXTOK));
  int b   = (int)(r / (PTOK + NXTOK));
  float v = x[tid];
  if (row < PTOK) memin[((size_t)(b * PTOK + row)) * DIMD + d] = v;
  else            xin[((size_t)(b * NXTOK + (row - PTOK))) * DIMD + d] = v;
}

__global__ __launch_bounds__(256) void concat_out_kernel(
    const float* __restrict__ mem, const float* __restrict__ xo, float* __restrict__ out)
{
  size_t tid = (size_t)blockIdx.x * 256 + threadIdx.x;
  const size_t total = (size_t)BATCH * (PTOK + NXTOK) * DIMD;
  if (tid >= total) return;
  int d = (int)(tid & (DIMD - 1));
  size_t r = tid >> 9;
  int row = (int)(r % (PTOK + NXTOK));
  int b   = (int)(r / (PTOK + NXTOK));
  out[tid] = (row < PTOK)
      ? mem[((size_t)(b * PTOK + row)) * DIMD + d]
      : xo[((size_t)(b * NXTOK + (row - PTOK))) * DIMD + d];
}

// Zp_sub (NSEQ*PTOK, 512) -> mean over V -> (B*PTOK, 512)
__global__ __launch_bounds__(256) void mtu_mean_kernel(
    const float* __restrict__ zp, float* __restrict__ mn)
{
  int tid = blockIdx.x * 256 + threadIdx.x;
  if (tid >= NMROWS * DIMD) return;
  int d = tid & (DIMD - 1);
  int p = (tid >> 9) % PTOK;
  int b = tid / (DIMD * PTOK);
  float s = 0.f;
  #pragma unroll
  for (int v = 0; v < 8; v++)
    s += zp[((size_t)((b * 8 + v) * PTOK + p)) * DIMD + d];
  mn[tid] = s * 0.125f;
}

// mem = mem + (mem + t)  ==  2*mem + t
__global__ __launch_bounds__(256) void mem_update_kernel(
    float* __restrict__ mem, const float* __restrict__ t, int n)
{
  int i = blockIdx.x * 256 + threadIdx.x;
  if (i < n) mem[i] = 2.0f * mem[i] + t[i];
}

// ---------------------------------------------------------------------------
// Host-side launcher
// ---------------------------------------------------------------------------
template<int BIAS, int ACT, int RES>
static void launch_gemm(const float* A, const float* W, const float* bias,
                        const float* res, float* C, int M, int N, int K, hipStream_t s)
{
  int mBlocks = ((M >> 4) + 7) / 8;
  int blocks  = mBlocks * (N >> 6);
  hipLaunchKernelGGL((gemm_f16wmma<BIAS, ACT, RES>), dim3(blocks), dim3(256), 0, s,
                     A, W, bias, res, C, M, N, K);
}

static void launch_ln(const float* X, const float* g, const float* b, float* Y,
                      int R, hipStream_t s)
{
  hipLaunchKernelGGL(layernorm512, dim3((R + 7) / 8), dim3(256), 0, s, X, g, b, Y, R);
}

extern "C" void kernel_launch(void* const* d_in, const int* in_sizes, int n_in,
                              void* d_out, int out_size, void* d_ws, size_t ws_size,
                              hipStream_t stream)
{
  (void)in_sizes; (void)n_in; (void)out_size; (void)ws_size;
  // --- input mapping (setup_inputs dict order, params depth-first) ---
  const float* x_in_full = (const float*)d_in[0];
  const float* ts        = (const float*)d_in[1];
  // mcx
  const float* m_lnq_g = (const float*)d_in[2];
  const float* m_lnq_b = (const float*)d_in[3];
  const float* m_lnkv_g= (const float*)d_in[4];
  const float* m_lnkv_b= (const float*)d_in[5];
  const float* m_Wq    = (const float*)d_in[6];
  const float* m_Wkv   = (const float*)d_in[7];
  const float* m_Wo    = (const float*)d_in[8];
  const float* m_bo    = (const float*)d_in[9];
  // xcm
  const float* c_lnq_g = (const float*)d_in[10];
  const float* c_lnq_b = (const float*)d_in[11];
  const float* c_lnkv_g= (const float*)d_in[12];
  const float* c_lnkv_b= (const float*)d_in[13];
  const float* c_Wq    = (const float*)d_in[14];
  const float* c_Wkv   = (const float*)d_in[15];
  const float* c_Wo    = (const float*)d_in[16];
  const float* c_bo    = (const float*)d_in[17];
  // mtu
  const float* prefix  = (const float*)d_in[18];
  const float* signal  = (const float*)d_in[19];
  const float* f_nu    = (const float*)d_in[20];
  const float* f_th    = (const float*)d_in[21];
  const float* f_gr    = (const float*)d_in[22];
  const float* f_gi    = (const float*)d_in[23];
  const float* b_nu    = (const float*)d_in[24];
  const float* b_th    = (const float*)d_in[25];
  const float* b_gr    = (const float*)d_in[26];
  const float* b_gi    = (const float*)d_in[27];
  const float* projW   = (const float*)d_in[28];
  const float* projB   = (const float*)d_in[29];
  const float* ln1_g   = (const float*)d_in[30];
  const float* ln1_b   = (const float*)d_in[31];
  // ffn_i
  const float* fi_ln_g = (const float*)d_in[32];
  const float* fi_ln_b = (const float*)d_in[33];
  const float* fi_W1   = (const float*)d_in[34];
  const float* fi_b1   = (const float*)d_in[35];
  const float* fi_W2   = (const float*)d_in[36];
  const float* fi_b2   = (const float*)d_in[37];
  // ffn_p
  const float* fp_ln_g = (const float*)d_in[38];
  const float* fp_ln_b = (const float*)d_in[39];
  const float* fp_W1   = (const float*)d_in[40];
  const float* fp_b1   = (const float*)d_in[41];
  const float* fp_W2   = (const float*)d_in[42];
  const float* fp_b2   = (const float*)d_in[43];

  float* outp = (float*)d_out;

  // --- workspace layout (floats), phased reuse ---
  float* w = (float*)d_ws;
  size_t off = 0;
  auto alloc = [&](size_t n){ float* p = w + off; off += n; return p; };
  float* BIGA  = alloc((size_t)NXROWS * 2048); // kvh_x | qh_x | h1
  float* B0    = alloc((size_t)NXROWS * 512);  // ln scratch / o_x
  float* XO    = alloc((size_t)NXROWS * 512);  // xo (live to the end)
  float* XIN   = alloc((size_t)NXROWS * 512);  // contiguous x[:,P:]
  float* MEMIN = alloc((size_t)NMROWS * 512);  // contiguous x[:,:P]
  float* SQN   = alloc((size_t)NMROWS * 512);
  float* SQH   = alloc((size_t)NMROWS * 512);
  float* SOM   = alloc((size_t)NMROWS * 512);
  float* SMEM  = alloc((size_t)NMROWS * 512);
  float* SKVN  = alloc((size_t)NMROWS * 512);
  float* SKVH  = alloc((size_t)NMROWS * 1024);
  float* YSUB  = alloc((size_t)NSEQ * PTOK * 1024);
  float* SZP   = alloc((size_t)NSEQ * PTOK * 512);
  float* SMEAN = alloc((size_t)NMROWS * 512);
  float* SLNT  = alloc((size_t)NMROWS * 512);
  float* SLN2  = alloc((size_t)NMROWS * 512);
  float* SH1P  = alloc((size_t)NMROWS * 2048);

  const size_t totalElems = (size_t)BATCH * (PTOK + NXTOK) * DIMD;
  const int elemBlocks = (int)((totalElems + 255) / 256);

  // split x -> memin, xin
  hipLaunchKernelGGL(split_x_kernel, dim3(elemBlocks), dim3(256), 0, stream,
                     x_in_full, MEMIN, XIN);

  // ---- Phase A: mcx (mem attends over x) ----
  launch_ln(MEMIN, m_lnq_g, m_lnq_b, SQN, NMROWS, stream);
  launch_ln(XIN,   m_lnkv_g, m_lnkv_b, B0, NXROWS, stream);
  launch_gemm<0,0,0>(SQN, m_Wq, nullptr, nullptr, SQH, NMROWS, 512, 512, stream);
  launch_gemm<0,0,0>(B0, m_Wkv, nullptr, nullptr, BIGA, NXROWS, 1024, 512, stream);
  hipLaunchKernelGGL(attn_qsmall, dim3(BATCH * HEADS * PTOK), dim3(256), 0, stream,
                     SQH, BIGA, SOM);
  launch_gemm<1,0,1>(SOM, m_Wo, m_bo, MEMIN, SMEM, NMROWS, 512, 512, stream);

  // ---- Phase B: xcm (x attends over mem) ----
  launch_ln(XIN,   c_lnq_g, c_lnq_b, B0, NXROWS, stream);
  launch_ln(MEMIN, c_lnkv_g, c_lnkv_b, SKVN, NMROWS, stream);
  launch_gemm<0,0,0>(B0, c_Wq, nullptr, nullptr, BIGA, NXROWS, 512, 512, stream); // qh_x
  launch_gemm<0,0,0>(SKVN, c_Wkv, nullptr, nullptr, SKVH, NMROWS, 1024, 512, stream);
  hipLaunchKernelGGL(attn_ksmall, dim3(NXTOK / 256, HEADS, BATCH), dim3(256), 0, stream,
                     BIGA, SKVH, B0);                                              // o_x
  launch_gemm<1,0,1>(B0, c_Wo, c_bo, XIN, XO, NXROWS, 512, 512, stream);

  // ---- Phase C: ffn_i on xo (residual) ----
  launch_ln(XO, fi_ln_g, fi_ln_b, B0, NXROWS, stream);
  launch_gemm<1,1,0>(B0, fi_W1, fi_b1, nullptr, BIGA, NXROWS, 2048, 512, stream);  // gelu
  launch_gemm<1,0,1>(BIGA, fi_W2, fi_b2, XO, XO, NXROWS, 512, 2048, stream);

  // ---- Phase D: MTU ----
  hipLaunchKernelGGL(lru_fwd_kernel, dim3(NSEQ * DIMD / 256), dim3(256), 0, stream,
                     SMEM, prefix, f_nu, f_th, f_gr, f_gi, YSUB);
  hipLaunchKernelGGL(lru_bwd_kernel, dim3(NSEQ * DIMD / 256), dim3(256), 0, stream,
                     SMEM, ts, prefix, signal, b_nu, b_th, b_gr, b_gi, YSUB);
  launch_gemm<1,0,0>(YSUB, projW, projB, nullptr, SZP, NSEQ * PTOK, 512, 1024, stream);
  hipLaunchKernelGGL(mtu_mean_kernel, dim3(NMROWS * DIMD / 256), dim3(256), 0, stream,
                     SZP, SMEAN);
  launch_ln(SMEAN, ln1_g, ln1_b, SLNT, NMROWS, stream);
  hipLaunchKernelGGL(mem_update_kernel, dim3(NMROWS * DIMD / 256), dim3(256), 0, stream,
                     SMEM, SLNT, NMROWS * DIMD);

  // ---- Phase E: ffn_p on mem (residual) ----
  launch_ln(SMEM, fp_ln_g, fp_ln_b, SLN2, NMROWS, stream);
  launch_gemm<1,1,0>(SLN2, fp_W1, fp_b1, nullptr, SH1P, NMROWS, 2048, 512, stream);
  launch_gemm<1,0,1>(SH1P, fp_W2, fp_b2, SMEM, SMEM, NMROWS, 512, 2048, stream);

  // ---- Phase F: concat(mem, xo) -> out ----
  hipLaunchKernelGGL(concat_out_kernel, dim3(elemBlocks), dim3(256), 0, stream,
                     SMEM, XO, outp);
}
